// CWTHead_14766097563874
// MI455X (gfx1250) — matrix operand order repaired and checked by reading, not compiled, not run on hardware
//
#include <hip/hip_runtime.h>
#include <math.h>

// ---------------------------------------------------------------------------
// CWT head on gfx1250: im2col GEMM via v_wmma_f32_16x16x32_bf16.
//   D[t, f'] = sum_k xp[t+k] * Wt[f'][k]     (A = signal Hankel, B = wavelets)
// f' is a permuted/padded filter axis (64) so real/imag of each frequency land
// in the same lane across accumulator tiles.
// Each wave owns 2 M-tiles (32 time rows) x 4 N-tiles (64 filters): the 4 B
// fragments loaded per K-step feed 8 WMMAs (1 ds_load_b128 per WMMA).
// ---------------------------------------------------------------------------

typedef __attribute__((ext_vector_type(16))) __bf16        v16bf;
typedef __attribute__((ext_vector_type(8)))  float         v8f;
typedef __attribute__((ext_vector_type(4)))  unsigned int  v4u;
typedef __attribute__((ext_vector_type(8)))  unsigned int  v8u;

#define T_LEN   2560
#define KW      815          // wavelet taps
#define PAD     407          // KW/2, reflect pad
#define KPAD    832          // 26 * 32
#define NF      25           // frequencies
#define MP      64           // padded/permuted filter count
#define WROW    840          // LDS row stride (halves): 1680B -> bank-conflict free
#define WSROW   832          // ws row stride (halves)
#define TILE_T  128          // time points per workgroup (32 per wave)
#define XP_LEN  960          // TILE_T + KPAD
#define WL_HALVES (MP * WROW)

// f32 -> bf16 round-to-nearest-even, pure integer
static __device__ __forceinline__ unsigned short f2bf(float f) {
    unsigned u = __builtin_bit_cast(unsigned, f);
    unsigned r = u + 0x7FFFu + ((u >> 16) & 1u);
    return (unsigned short)(r >> 16);
}

struct B2 { v4u lo, hi; };   // 32B container for a v16bf fragment

// ---------------------------------------------------------------------------
// Prep: permute wavelets (50 x 815 f32) into ws as bf16 [64][832]:
//   f' in [ 0,16): real of freq f'        f' in [16,32): imag of freq f'-16
//   f' in [32,48): real of freq 16+f'-32  f' in [48,64): imag of freq 16+f'-48
// Out-of-range freqs / taps are zero (contribute nothing to the GEMM).
// ---------------------------------------------------------------------------
__global__ void cwt_prep_wavelets(const float* __restrict__ w,
                                  unsigned short* __restrict__ ws) {
    int idx = blockIdx.x * blockDim.x + threadIdx.x;
    if (idx >= MP * KPAD) return;
    int fp = idx / KPAD;
    int k  = idx % KPAD;
    int grp  = fp >> 4;
    int fl   = fp & 15;
    int f    = (grp & 2) ? (16 + fl) : fl;
    int part = grp & 1;                    // 0 = real row, 1 = imag row
    float v = 0.0f;
    if (f < NF && k < KW) v = w[(2 * f + part) * KW + k];
    ws[fp * WSROW + k] = f2bf(v);
}

// magnitude/phase + scatter into the (B,2,F,7,5,T) montage
static __device__ __forceinline__ void emit_out(
    float* __restrict__ out, int b, int f, int row, int col, int t, bool dup,
    float re, float im, const float* __restrict__ pshift,
    const float* __restrict__ pscale, float phsc)
{
    float mag = log1pf(sqrtf(fmaf(re, re, fmaf(im, im, 1e-8f))));
    mag = (mag + pshift[f]) * pscale[f];
    float ph = atan2f(im, re) * phsc;
    const size_t pstride = (size_t)NF * 35 * T_LEN;          // p axis stride
    size_t o = (((size_t)b * 2 * NF + f) * 35 + row * 5 + col) * (size_t)T_LEN + t;
    out[o]           = mag;
    out[o + pstride] = ph;
    if (dup) {  // montage row 0 duplicates of row-1 cols {0,2,4} (ch 2,4,6)
        out[o - 5 * T_LEN]           = mag;
        out[o + pstride - 5 * T_LEN] = ph;
    }
}

__global__ __launch_bounds__(128) void cwt_wmma(
    const float* __restrict__ x,
    const unsigned short* __restrict__ ws,
    const float* __restrict__ pshift,
    const float* __restrict__ pscale,
    const float* __restrict__ phsc_p,
    const int*   __restrict__ rows,
    const int*   __restrict__ cols,
    float*       __restrict__ out)
{
    extern __shared__ unsigned short smem[];
    unsigned short* wl = smem;              // [64][840] bf16 wavelets
    unsigned short* xp = smem + WL_HALVES;  // [960] bf16 signal window

    const int tid = threadIdx.x;
    const int t0  = blockIdx.x * TILE_T;
    const int bc  = blockIdx.y;             // b*32 + ch
    const int ch  = bc & 31;
    const int b   = bc >> 5;

    // --- stage wavelets ws -> LDS (re-stride 832 -> 840 halves), 16B chunks
    for (int c = tid; c < MP * (KPAD / 8); c += 128) {
        int r = c / (KPAD / 8);
        int o = (c % (KPAD / 8)) * 8;
        *(v4u*)(wl + r * WROW + o) = *(const v4u*)(ws + r * WSROW + o);
    }
    // --- stage reflect-padded signal window -> LDS bf16 (7.5 elems/thread)
    const float* xrow = x + (size_t)bc * T_LEN;
    for (int i = tid; i < XP_LEN; i += 128) {
        int s = t0 + i - PAD;
        if (s < 0) s = -s;
        if (s >= T_LEN) s = 2 * T_LEN - 2 - s;
        xp[i] = f2bf(xrow[s]);
    }
    __syncthreads();

    const int wave  = tid >> 5;
    const int lane  = tid & 31;
    const int lhalf = lane >> 4;    // 0: lanes 0-15, 1: lanes 16-31
    const int l16   = lane & 15;

    v8f acc[2][4];
    #pragma unroll
    for (int mt = 0; mt < 2; ++mt)
        #pragma unroll
        for (int nt = 0; nt < 4; ++nt)
            acc[mt][nt] = v8f{};

    // A-fragment bases: wave owns times [t0 + 32*wave, +32), M-tile mt adds 16
    const int abase = wave * 32 + l16;
    const int boff  = l16 * WROW + lhalf * 8;    // B: N = filter = nt*16 + l16

    for (int kk = 0; kk < KPAD; kk += 32) {
        // A (16x32 bf16, ISA layout): lanes 0-15 hold K kk+0..7 / kk+16..23,
        // lanes 16-31 hold K kk+8..15 / kk+24..31. Per-lane contiguous halves.
        v16bf a[2];
        #pragma unroll
        for (int mt = 0; mt < 2; ++mt) {
            int a0 = abase + mt * 16 + kk + lhalf * 8;
            v8u araw;
            #pragma unroll
            for (int h = 0; h < 4; ++h) {
                araw[h]     = (unsigned)xp[a0 + 2*h]      | ((unsigned)xp[a0 + 2*h + 1]  << 16);
                araw[4 + h] = (unsigned)xp[a0 + 16 + 2*h] | ((unsigned)xp[a0 + 17 + 2*h] << 16);
            }
            a[mt] = __builtin_bit_cast(v16bf, araw);
        }

        // B (32x16 bf16): lanes 0-15 K kk+0..7 / kk+16..23, lanes 16-31 K kk+8..15 / kk+24..31,
        // two aligned ds_load_b128 per tile; each B feeds both M-tiles.
        #pragma unroll
        for (int nt = 0; nt < 4; ++nt) {
            const v4u* bp = (const v4u*)(wl + nt * 16 * WROW + boff + kk);
            B2 braw = { bp[0], bp[2] };
            v16bf bm = __builtin_bit_cast(v16bf, braw);
            #pragma unroll
            for (int mt = 0; mt < 2; ++mt)
                acc[mt][nt] = __builtin_amdgcn_wmma_f32_16x16x32_bf16(
                    false, a[mt], false, bm, (short)0, acc[mt][nt], false, false);
        }
    }

    // --- epilogue: acc[mt][0/1] = re/im of freq l16, acc[mt][2/3] = 16+l16.
    // C layout: VGPR j -> t_local = 32*wave + 16*mt + j + 8*lhalf, N = l16.
    const float phsc = *phsc_p;
    const int row = rows[ch];
    const int col = cols[ch];
    const bool dup = (row == 1) && ((col & 1) == 0);   // ch 2,4,6 feed row 0

    #pragma unroll
    for (int mt = 0; mt < 2; ++mt) {
        #pragma unroll
        for (int j = 0; j < 8; ++j) {
            int t = t0 + wave * 32 + mt * 16 + lhalf * 8 + j;
            emit_out(out, b, l16, row, col, t, dup,
                     acc[mt][0][j], acc[mt][1][j], pshift, pscale, phsc);
            if (l16 < NF - 16)  // freqs 16..24 only
                emit_out(out, b, 16 + l16, row, col, t, dup,
                         acc[mt][2][j], acc[mt][3][j], pshift, pscale, phsc);
        }
    }
}

extern "C" void kernel_launch(void* const* d_in, const int* in_sizes, int n_in,
                              void* d_out, int out_size, void* d_ws, size_t ws_size,
                              hipStream_t stream) {
    const float* x      = (const float*)d_in[0];
    const float* w      = (const float*)d_in[1];
    const float* pshift = (const float*)d_in[2];
    const float* pscale = (const float*)d_in[3];
    const float* phsc   = (const float*)d_in[4];
    const int*   rows   = (const int*)d_in[5];
    const int*   cols   = (const int*)d_in[6];
    unsigned short* ws  = (unsigned short*)d_ws;   // 64*832*2 = 106,496 B
    float* out          = (float*)d_out;

    cwt_prep_wavelets<<<(MP * KPAD + 255) / 256, 256, 0, stream>>>(w, ws);

    dim3 grid(T_LEN / TILE_T, 8 * 32);             // 20 time tiles x 256 signals
    size_t shmem = (size_t)(WL_HALVES + XP_LEN) * sizeof(unsigned short); // ~109 KB
    cwt_wmma<<<grid, 128, shmem, stream>>>(x, ws, pshift, pscale, phsc,
                                           rows, cols, out);
}
